// Attn_30279519437427
// MI455X (gfx1250) — matrix operand order repaired and checked by reading, not compiled, not run on hardware
//
#include <hip/hip_runtime.h>

typedef __bf16 v16bf __attribute__((ext_vector_type(16)));
typedef __bf16 v8bf  __attribute__((ext_vector_type(8)));
typedef float  v8f   __attribute__((ext_vector_type(8)));

#define B_   2
#define T_   2048
#define D_   2048
#define H_   16
#define HKV_ 4
#define DH_  128
#define BT_  (B_ * T_)

// ---------- helpers ----------
__device__ __forceinline__ v16bf ld_frag16(const __bf16* lo, const __bf16* hi) {
  v8bf a = *(const v8bf*)lo;
  v8bf b = *(const v8bf*)hi;
  return __builtin_shufflevector(a, b, 0,1,2,3,4,5,6,7,8,9,10,11,12,13,14,15);
}

__device__ __forceinline__ v8f wmma_bf16(v16bf a, v16bf b, v8f c) {
  return __builtin_amdgcn_wmma_f32_16x16x32_bf16(false, a, false, b, (short)0, c, false, false);
}

// async copy: 16B per lane, global -> LDS, no VGPR round-trip (ASYNCcnt path)
__device__ __forceinline__ void async_copy_b128(__bf16* lds_dst, const __bf16* gsrc) {
  unsigned loff = (unsigned)(unsigned long long)lds_dst;  // low 32b of generic ptr = LDS offset
  asm volatile("global_load_async_to_lds_b128 %0, %1, off"
               :: "v"(loff), "v"(gsrc) : "memory");
}
__device__ __forceinline__ void wait_async0() {
  asm volatile("s_wait_asynccnt 0x0" ::: "memory");
}

// ---------- fp32 -> bf16 convert ----------
__global__ void cvt_bf16(const float* __restrict__ in, __bf16* __restrict__ out, int n) {
  int i = blockIdx.x * blockDim.x + threadIdx.x;
  if (i < n) out[i] = (__bf16)in[i];
}

// ---------- GEMM: C(MxN) = A(MxK) @ B(NxK)^T, bf16 in, fp32 acc ----------
template <typename OutT>
__global__ __launch_bounds__(256)
void gemm_xwt(const __bf16* __restrict__ A, const __bf16* __restrict__ Bw,
              OutT* __restrict__ C, int M, int N, int K) {
  __shared__ __bf16 As[128][40];
  __shared__ __bf16 Bs[128][40];
  const int tid  = threadIdx.x;
  const int lane = tid & 31, w = tid >> 5;
  const int wm = w >> 1, wn = w & 1;
  const int lh = lane >> 4, ln = lane & 15;
  const int m0 = blockIdx.y << 7, n0 = blockIdx.x << 7;

  v8f acc[2][4];
#pragma unroll
  for (int i = 0; i < 2; ++i)
#pragma unroll
    for (int j = 0; j < 4; ++j) acc[i][j] = (v8f){0.f,0.f,0.f,0.f,0.f,0.f,0.f,0.f};

  const int nkt = K >> 5;
  for (int kt = 0; kt < nkt; ++kt) {
    const int k0 = kt << 5;
#pragma unroll
    for (int i = 0; i < 2; ++i) {
      int c   = tid + (i << 8);       // 0..511 : 16B chunks
      int r   = c >> 2;               // 4 chunks per 32-col row
      int off = (c & 3) << 3;
      async_copy_b128(&As[r][off], A  + (size_t)(m0 + r) * K + k0 + off);
      async_copy_b128(&Bs[r][off], Bw + (size_t)(n0 + r) * K + k0 + off);
      if (kt + 1 < nkt) {
        __builtin_prefetch(A  + (size_t)(m0 + r) * K + k0 + 32 + off, 0, 1);
        __builtin_prefetch(Bw + (size_t)(n0 + r) * K + k0 + 32 + off, 0, 1);
      }
    }
    wait_async0();
    __syncthreads();

    v16bf af[2], bf[4];
    const int kbA = lh << 3;          // A: K groups of 8 per lane-half
    const int kbB = lh << 3;          // B: K groups of 8 per lane-half
#pragma unroll
    for (int mi = 0; mi < 2; ++mi) {
      const __bf16* p = &As[wm * 32 + mi * 16 + ln][kbA];
      af[mi] = ld_frag16(p, p + 16);
    }
#pragma unroll
    for (int ni = 0; ni < 4; ++ni) {
      const __bf16* p = &Bs[wn * 64 + ni * 16 + ln][kbB];
      bf[ni] = ld_frag16(p, p + 16);
    }
#pragma unroll
    for (int mi = 0; mi < 2; ++mi)
#pragma unroll
      for (int ni = 0; ni < 4; ++ni)
        acc[mi][ni] = wmma_bf16(af[mi], bf[ni], acc[mi][ni]);
    __syncthreads();
  }

#pragma unroll
  for (int mi = 0; mi < 2; ++mi)
#pragma unroll
    for (int ni = 0; ni < 4; ++ni)
#pragma unroll
      for (int r = 0; r < 8; ++r) {
        int row = m0 + wm * 32 + mi * 16 + r + (lh << 3);
        int col = n0 + wn * 64 + ni * 16 + ln;
        C[(size_t)row * N + col] = (OutT)acc[mi][ni][r];
      }
}

// ---------- fused RMS-norm + ortho-rope (in place, one wave per (row, head)) ----------
__global__ __launch_bounds__(256)
void rmsrope_kernel(__bf16* __restrict__ buf, const float* __restrict__ cosp,
                    const float* __restrict__ sinp, int nheads) {
  int gw   = blockIdx.x * 8 + (threadIdx.x >> 5);
  int lane = threadIdx.x & 31;
  int row  = gw / nheads;
  int h    = gw - row * nheads;
  int t    = row & (T_ - 1);
  __bf16* p = buf + (size_t)row * (nheads * DH_) + h * DH_;

  float x0a = (float)p[2 * lane],      x1a = (float)p[2 * lane + 1];
  float x0b = (float)p[2 * lane + 64], x1b = (float)p[2 * lane + 65];
  float ss = x0a * x0a + x1a * x1a + x0b * x0b + x1b * x1b;
#pragma unroll
  for (int m = 1; m < 32; m <<= 1) ss += __shfl_xor(ss, m, 32);
  float rn = rsqrtf(ss * (1.f / DH_) + 1e-6f);
  float ca = cosp[t * 64 + lane],      sa = sinp[t * 64 + lane];
  float cb = cosp[t * 64 + 32 + lane], sb = sinp[t * 64 + 32 + lane];
  x0a *= rn; x1a *= rn; x0b *= rn; x1b *= rn;
  p[lane]      = (__bf16)(x0a * ca - x1a * sa);
  p[64 + lane] = (__bf16)(x1a * ca + x0a * sa);
  p[32 + lane] = (__bf16)(x0b * cb - x1b * sb);
  p[96 + lane] = (__bf16)(x1b * cb + x0b * sb);
}

// ---------- v = v + lambda*(v1 - v) ----------
__global__ void vmix_kernel(__bf16* __restrict__ vbuf, const float* __restrict__ v1,
                            const float* __restrict__ vlamb) {
  int i = blockIdx.x * blockDim.x + threadIdx.x;  // BT*512
  int row = i >> 9, c = i & 511;
  int hk = c >> 7, d = c & 127;
  int bb = row / T_, t = row & (T_ - 1);
  float lam = *vlamb;
  float v   = (float)vbuf[i];
  float v1v = v1[(((size_t)bb * HKV_ + hk) * T_ + t) * DH_ + d];
  vbuf[i] = (__bf16)(v + lam * (v1v - v));
}

// ---------- gates = sigmoid(x[:, :16] @ gate_w^T) ----------
__global__ void gate_kernel(const float* __restrict__ x, const float* __restrict__ gw,
                            float* __restrict__ gates) {
  int i = blockIdx.x * blockDim.x + threadIdx.x;  // BT*16
  int row = i >> 4, h = i & 15;
  const float* xr = x + (size_t)row * D_;
  float s = 0.f;
#pragma unroll
  for (int j = 0; j < 16; ++j) s += xr[j] * gw[h * 16 + j];
  gates[i] = 1.f / (1.f + __expf(-s));
}

// ---------- causal GQA flash attention, 128 q rows / block, WMMA bf16 ----------
__global__ __launch_bounds__(256)
void flash_attn(const __bf16* __restrict__ qb, const __bf16* __restrict__ kb,
                const __bf16* __restrict__ vb, const float* __restrict__ gates,
                __bf16* __restrict__ yb) {
  __shared__ __bf16 Ks[32][136];      // K tile, row-major [krow][dh]
  __shared__ __bf16 Vt[128][40];      // V tile transposed [dh][krow]
  __shared__ __bf16 Ps[8][16][40];    // per-wave P
  __shared__ __bf16 Ys[8][16][136];   // per-wave output staging

  const int tid = threadIdx.x, lane = tid & 31, w = tid >> 5;
  const int lh = lane >> 4, ln = lane & 15;
  const int q0 = blockIdx.x << 7;
  const int h  = blockIdx.y;
  const int b  = blockIdx.z;
  const int hk = h >> 2;
  const size_t rowbase = (size_t)b * T_;

  // Q fragments for this wave's 16 rows (A layout: row = lane%16)
  v16bf qf[4];
  const __bf16* qrow = qb + (rowbase + q0 + w * 16 + ln) * (size_t)D_ + h * DH_;
#pragma unroll
  for (int s = 0; s < 4; ++s) {
    const __bf16* p = qrow + s * 32 + (lh << 3);
    qf[s] = ld_frag16(p, p + 16);
  }

  v8f o[8];
#pragma unroll
  for (int i = 0; i < 8; ++i) o[i] = (v8f){0.f,0.f,0.f,0.f,0.f,0.f,0.f,0.f};
  float mrow[8], lrow[8];
#pragma unroll
  for (int r = 0; r < 8; ++r) { mrow[r] = -1e30f; lrow[r] = 0.f; }

  const float scale = 0.08838834764831845f;  // DH^-0.5
  const int nkt = (q0 >> 5) + 4;
  for (int kt = 0; kt < nkt; ++kt) {
    const int k0 = kt << 5;
#pragma unroll
    for (int i = 0; i < 2; ++i) {            // K tile: 512 x 16B chunks, async -> LDS
      int c = tid + (i << 8);
      int r = c >> 4;
      int off = (c & 15) << 3;
      async_copy_b128(&Ks[r][off],
                      kb + (rowbase + k0 + r) * (size_t)(HKV_ * DH_) + hk * DH_ + off);
    }
#pragma unroll
    for (int i = 0; i < 16; ++i) {           // V tile transposed (scalar path)
      int e = tid + (i << 8);
      int r = e >> 7, cC = e & 127;
      Vt[cC][r] = vb[(rowbase + k0 + r) * (size_t)(HKV_ * DH_) + hk * DH_ + cC];
    }
    wait_async0();
    __syncthreads();

    // S = Q @ K^T : two 16x16 N-tiles, K over DH in 4 steps
    v8f s0 = (v8f){0.f,0.f,0.f,0.f,0.f,0.f,0.f,0.f};
    v8f s1 = s0;
#pragma unroll
    for (int s = 0; s < 4; ++s) {
      const __bf16* p0 = &Ks[ln][(s << 5) + (lh << 3)];
      const __bf16* p1 = &Ks[16 + ln][(s << 5) + (lh << 3)];
      v16bf kf0 = ld_frag16(p0, p0 + 16);
      v16bf kf1 = ld_frag16(p1, p1 + 16);
      s0 = wmma_bf16(qf[s], kf0, s0);
      s1 = wmma_bf16(qf[s], kf1, s1);
    }

    // online softmax (row reductions inside 16-lane halves match C-layout)
#pragma unroll
    for (int r = 0; r < 8; ++r) {
      int row = q0 + w * 16 + r + (lh << 3);
      float a0 = (k0 + ln      <= row) ? s0[r] * scale : -1e30f;
      float a1 = (k0 + 16 + ln <= row) ? s1[r] * scale : -1e30f;
      float mx = fmaxf(a0, a1);
#pragma unroll
      for (int msk = 1; msk < 16; msk <<= 1) mx = fmaxf(mx, __shfl_xor(mx, msk, 32));
      float mnew  = fmaxf(mrow[r], mx);
      float alpha = __expf(mrow[r] - mnew);
      float p0 = __expf(a0 - mnew);
      float p1 = __expf(a1 - mnew);
      float ps = p0 + p1;
#pragma unroll
      for (int msk = 1; msk < 16; msk <<= 1) ps += __shfl_xor(ps, msk, 32);
      lrow[r] = lrow[r] * alpha + ps;
      mrow[r] = mnew;
#pragma unroll
      for (int nt = 0; nt < 8; ++nt) o[nt][r] *= alpha;
      Ps[w][r + (lh << 3)][ln]      = (__bf16)p0;
      Ps[w][r + (lh << 3)][16 + ln] = (__bf16)p1;
    }
    asm volatile("s_wait_dscnt 0" ::: "memory");  // per-wave P visible in LDS

    // O += P @ V  (P as A-fragment from LDS, V^T rows give B-fragments)
    const __bf16* pp = &Ps[w][ln][lh << 3];
    v16bf pf = ld_frag16(pp, pp + 16);
#pragma unroll
    for (int nt = 0; nt < 8; ++nt) {
      const __bf16* vp = &Vt[nt * 16 + ln][lh << 3];
      v16bf vf = ld_frag16(vp, vp + 16);
      o[nt] = wmma_bf16(pf, vf, o[nt]);
    }
    __syncthreads();
  }

  // epilogue: 1/l, gate, stage in LDS, vectorized store
  float gr[8], rl[8];
#pragma unroll
  for (int r = 0; r < 8; ++r) {
    int row = q0 + w * 16 + r + (lh << 3);
    gr[r] = gates[(rowbase + row) * H_ + h];
    rl[r] = 1.f / lrow[r];
  }
#pragma unroll
  for (int nt = 0; nt < 8; ++nt)
#pragma unroll
    for (int r = 0; r < 8; ++r)
      Ys[w][r + (lh << 3)][nt * 16 + ln] = (__bf16)(o[nt][r] * rl[r] * gr[r]);
  asm volatile("s_wait_dscnt 0" ::: "memory");
#pragma unroll
  for (int j = 0; j < 8; ++j) {
    int cc = lane + (j << 5);
    int r = cc >> 4;
    int off = (cc & 15) << 3;
    *(v8bf*)(yb + (rowbase + q0 + w * 16 + r) * (size_t)D_ + h * DH_ + off) =
        *(const v8bf*)&Ys[w][r][off];
  }
}

// ---------- host orchestration ----------
extern "C" void kernel_launch(void* const* d_in, const int* in_sizes, int n_in,
                              void* d_out, int out_size, void* d_ws, size_t ws_size,
                              hipStream_t stream) {
  (void)in_sizes; (void)n_in; (void)out_size; (void)ws_size;
  const float* x    = (const float*)d_in[0];
  const float* cosp = (const float*)d_in[2];
  const float* sinp = (const float*)d_in[3];
  const float* v1   = (const float*)d_in[4];
  const float* Wq   = (const float*)d_in[5];
  const float* Wk   = (const float*)d_in[6];
  const float* Wv   = (const float*)d_in[7];
  const float* Wo   = (const float*)d_in[8];
  const float* gw   = (const float*)d_in[9];
  const float* vl   = (const float*)d_in[10];

  char* ws = (char*)d_ws;
  size_t off = 0;
  auto alloc = [&](size_t bytes) -> void* {
    void* p = ws + off; off += (bytes + 255) & ~(size_t)255; return p;
  };
  __bf16* xb    = (__bf16*)alloc((size_t)BT_ * D_ * 2);
  __bf16* Wqb   = (__bf16*)alloc((size_t)(H_ * DH_) * D_ * 2);
  __bf16* Wkb   = (__bf16*)alloc((size_t)(HKV_ * DH_) * D_ * 2);
  __bf16* Wvb   = (__bf16*)alloc((size_t)(HKV_ * DH_) * D_ * 2);
  __bf16* Wob   = (__bf16*)alloc((size_t)D_ * D_ * 2);
  __bf16* qbuf  = (__bf16*)alloc((size_t)BT_ * (H_ * DH_) * 2);
  __bf16* kbuf  = (__bf16*)alloc((size_t)BT_ * (HKV_ * DH_) * 2);
  __bf16* vbuf  = (__bf16*)alloc((size_t)BT_ * (HKV_ * DH_) * 2);
  __bf16* ybuf  = (__bf16*)alloc((size_t)BT_ * D_ * 2);
  float*  gates = (float*)alloc((size_t)BT_ * H_ * 4);

  // 1) fp32 -> bf16
  auto cvt = [&](const float* src, __bf16* dst, int n) {
    cvt_bf16<<<(n + 255) / 256, 256, 0, stream>>>(src, dst, n);
  };
  cvt(x,  xb,  BT_ * D_);
  cvt(Wq, Wqb, H_ * DH_ * D_);
  cvt(Wk, Wkb, HKV_ * DH_ * D_);
  cvt(Wv, Wvb, HKV_ * DH_ * D_);
  cvt(Wo, Wob, D_ * D_);

  // 2) QKV projections
  gemm_xwt<__bf16><<<dim3((H_ * DH_) / 128, BT_ / 128), 256, 0, stream>>>(
      xb, Wqb, qbuf, BT_, H_ * DH_, D_);
  gemm_xwt<__bf16><<<dim3((HKV_ * DH_) / 128, BT_ / 128), 256, 0, stream>>>(
      xb, Wkb, kbuf, BT_, HKV_ * DH_, D_);
  gemm_xwt<__bf16><<<dim3((HKV_ * DH_) / 128, BT_ / 128), 256, 0, stream>>>(
      xb, Wvb, vbuf, BT_, HKV_ * DH_, D_);

  // 3) norm+rope (q, k), v mix, gates
  rmsrope_kernel<<<(BT_ * H_) / 8, 256, 0, stream>>>(qbuf, cosp, sinp, H_);
  rmsrope_kernel<<<(BT_ * HKV_) / 8, 256, 0, stream>>>(kbuf, cosp, sinp, HKV_);
  vmix_kernel<<<(BT_ * HKV_ * DH_) / 256, 256, 0, stream>>>(vbuf, v1, vl);
  gate_kernel<<<(BT_ * H_) / 256, 256, 0, stream>>>(x, gw, gates);

  // 4) flash attention (causal, GQA), gated bf16 output
  flash_attn<<<dim3(T_ / 128, H_, B_), 256, 0, stream>>>(qbuf, kbuf, vbuf, gates, ybuf);

  // 5) output projection -> fp32 d_out
  gemm_xwt<float><<<dim3(D_ / 128, BT_ / 128), 256, 0, stream>>>(
      ybuf, Wob, (float*)d_out, BT_, D_, D_);

  // 6) second tuple element: v1 passthrough
  hipMemcpyAsync((float*)d_out + (size_t)BT_ * D_, v1,
                 (size_t)B_ * HKV_ * T_ * DH_ * sizeof(float),
                 hipMemcpyDeviceToDevice, stream);
}
